// PatchVQVAETransformer_10239202034435
// MI455X (gfx1250) — compile-verified, matrix operand order checked by reading, not run on hardware
//
#include <hip/hip_runtime.h>

typedef __attribute__((ext_vector_type(16))) __bf16 v16bf;
typedef __attribute__((ext_vector_type(8)))  float  v8f;

#define MTOT 16384   // B*P tokens
#define DDIM 512
#define NQKV 1536
#define DFFN 2048
#define NCODE 1024
#define PWAVES 4

__device__ __forceinline__ v8f wmma_bf16(v16bf a, v16bf b, v8f c) {
  // D = A(16x32) x B(32x16) + C, bf16 in / f32 out
  return __builtin_amdgcn_wmma_f32_16x16x32_bf16(false, a, false, b, (short)0, c,
                                                 false, false);
}

__device__ __forceinline__ float gelu_exact(float x) {
  return 0.5f * x * (1.0f + erff(x * 0.70710678118654752f));
}

// ---------------------------------------------------------------------------
// fp32 -> bf16 conversion kernels (weights / codebook)
// ---------------------------------------------------------------------------
__global__ __launch_bounds__(256) void cvt_kernel(const float* __restrict__ in,
                                                  __bf16* __restrict__ out, int n) {
  const int i = blockIdx.x * 256 + threadIdx.x;
  if (i < n) out[i] = (__bf16)in[i];
}

// [L][R][Cc] fp32 -> [L][Cc][R] bf16
__global__ __launch_bounds__(256) void cvt_tr_kernel(const float* __restrict__ in,
                                                     __bf16* __restrict__ out,
                                                     int R, int Cc, int total) {
  const int i = blockIdx.x * 256 + threadIdx.x;
  if (i >= total) return;
  const int c  = i % Cc;
  const int rr = (i / Cc) % R;
  const int ll = i / (Cc * R);
  out[((size_t)ll * Cc + c) * R + rr] = (__bf16)in[i];
}

// cc[k] = sum_d cb[k][d]^2   (one wave per row)
__global__ __launch_bounds__(256) void cc_kernel(const float* __restrict__ cb,
                                                 float* __restrict__ cc) {
  const int l = threadIdx.x & 31;
  const int w = threadIdx.x >> 5;
  const int row = blockIdx.x * 8 + w;
  float s = 0.0f;
#pragma unroll
  for (int i = 0; i < 16; ++i) {
    const float v = cb[(size_t)row * DDIM + l + 32 * i];
    s += v * v;
  }
#pragma unroll
  for (int msk = 1; msk <= 16; msk <<= 1) s += __shfl_xor(s, msk, 32);
  if (l == 0) cc[row] = s;
}

// ---------------------------------------------------------------------------
// Patch cross-attention encoder: one wave per patch, fp32 VALU (C=32 is tiny).
// Produces z (fp32) and z (bf16).
// ---------------------------------------------------------------------------
__global__ __launch_bounds__(PWAVES * 32) void patch_encode_kernel(
    const float* __restrict__ x,
    const float* __restrict__ lq_g, const float* __restrict__ pos_g,
    const float* __restrict__ win, const float* __restrict__ bin,
    const float* __restrict__ wout, const float* __restrict__ bout,
    const float* __restrict__ ln1g, const float* __restrict__ ln1b,
    const float* __restrict__ ln2g, const float* __restrict__ ln2b,
    const float* __restrict__ w1, const float* __restrict__ b1,
    const float* __restrict__ w2, const float* __restrict__ b2,
    float* __restrict__ z, __bf16* __restrict__ zb) {
  __shared__ float lq_s[512];
  __shared__ float pos_s[512];
  __shared__ float buf[PWAVES][3584];
  __shared__ float mrow[PWAVES][16];
  __shared__ float vrow[PWAVES][16];

  const int tid = threadIdx.x;
  for (int i = tid; i < 512; i += PWAVES * 32) {
    lq_s[i] = lq_g[i];
    pos_s[i] = pos_g[i];
  }

  const int w = tid >> 5;
  const int j = tid & 31;
  float* xp = buf[w];         // [16][32] raw patch (kept for final residual)
  float* xs = buf[w] + 512;   // x + pos
  float* qa = buf[w] + 1024;  // q, then attention output
  float* kx = buf[w] + 1536;  // k, then xn
  float* vx = buf[w] + 2048;  // v, then xo pre-LN
  float* hb = buf[w] + 2560;  // [16][64] FFN hidden

  const size_t p = (size_t)blockIdx.x * PWAVES + w;
  const float* xrow = x + p * 512;
#pragma unroll
  for (int i = 0; i < 16; i += 4) {
    float4 t4 = *(const float4*)(xrow + j * 16 + i);
    xp[j * 16 + i + 0] = t4.x;
    xp[j * 16 + i + 1] = t4.y;
    xp[j * 16 + i + 2] = t4.z;
    xp[j * 16 + i + 3] = t4.w;
  }
  __syncthreads();
#pragma unroll
  for (int i = 0; i < 16; ++i) xs[j * 16 + i] = xp[j * 16 + i] + pos_s[j * 16 + i];
  __syncthreads();

  // ---- Q/K/V projections: lane j computes column j ----
  {
    float wr[32];
#pragma unroll
    for (int c = 0; c < 32; c += 4) {
      float4 t4 = *(const float4*)(win + j * 32 + c);
      wr[c] = t4.x; wr[c + 1] = t4.y; wr[c + 2] = t4.z; wr[c + 3] = t4.w;
    }
    const float bq = bin[j];
    for (int t = 0; t < 16; ++t) {
      float s = bq;
#pragma unroll
      for (int c = 0; c < 32; ++c) s += lq_s[t * 32 + c] * wr[c];
      qa[t * 32 + j] = s;
    }
  }
  {
    float wr[32];
#pragma unroll
    for (int c = 0; c < 32; c += 4) {
      float4 t4 = *(const float4*)(win + (32 + j) * 32 + c);
      wr[c] = t4.x; wr[c + 1] = t4.y; wr[c + 2] = t4.z; wr[c + 3] = t4.w;
    }
    const float bk = bin[32 + j];
    for (int t = 0; t < 16; ++t) {
      float s = bk;
#pragma unroll
      for (int c = 0; c < 32; ++c) s += xs[t * 32 + c] * wr[c];
      kx[t * 32 + j] = s;
    }
  }
  {
    float wr[32];
#pragma unroll
    for (int c = 0; c < 32; c += 4) {
      float4 t4 = *(const float4*)(win + (64 + j) * 32 + c);
      wr[c] = t4.x; wr[c + 1] = t4.y; wr[c + 2] = t4.z; wr[c + 3] = t4.w;
    }
    const float bv = bin[64 + j];
    for (int t = 0; t < 16; ++t) {
      float s = bv;
#pragma unroll
      for (int c = 0; c < 32; ++c) s += xs[t * 32 + c] * wr[c];
      vx[t * 32 + j] = s;
    }
  }
  __syncthreads();

  // ---- attention: lane handles (query t = j%16, head h), two heads per lane
  const float iss = 0.35355339059327373f;  // 1/sqrt(8)
#pragma unroll
  for (int pr = 0; pr < 2; ++pr) {
    const int t = j & 15;
    const int h = (j >> 4) + pr * 2;
    float sc[16];
    float mx = -3.0e38f;
#pragma unroll
    for (int u = 0; u < 16; ++u) {
      float s = 0.0f;
#pragma unroll
      for (int d = 0; d < 8; ++d)
        s += qa[t * 32 + h * 8 + d] * kx[u * 32 + h * 8 + d];
      s *= iss;
      sc[u] = s;
      mx = fmaxf(mx, s);
    }
    float sum = 0.0f;
#pragma unroll
    for (int u = 0; u < 16; ++u) { sc[u] = __expf(sc[u] - mx); sum += sc[u]; }
    const float inv = 1.0f / sum;
    float o[8] = {0, 0, 0, 0, 0, 0, 0, 0};
#pragma unroll
    for (int u = 0; u < 16; ++u) {
      const float pu = sc[u] * inv;
#pragma unroll
      for (int d = 0; d < 8; ++d) o[d] += pu * vx[u * 32 + h * 8 + d];
    }
#pragma unroll
    for (int d = 0; d < 8; ++d) qa[t * 32 + h * 8 + d] = o[d];  // disjoint cols
  }
  __syncthreads();

  // ---- out-proj + residual(lq) + LN1 -> kx holds xn ----
  {
    float wr[32];
#pragma unroll
    for (int c = 0; c < 32; c += 4) {
      float4 t4 = *(const float4*)(wout + j * 32 + c);
      wr[c] = t4.x; wr[c + 1] = t4.y; wr[c + 2] = t4.z; wr[c + 3] = t4.w;
    }
    const float bo = bout[j];
    for (int t = 0; t < 16; ++t) {
      float s = bo;
#pragma unroll
      for (int c = 0; c < 32; ++c) s += qa[t * 32 + c] * wr[c];
      kx[t * 32 + j] = s + lq_s[t * 32 + j];
    }
  }
  __syncthreads();
  if (j < 16) {
    float s = 0.0f;
    for (int c = 0; c < 32; ++c) s += kx[j * 32 + c];
    const float mn = s * (1.0f / 32.0f);
    float vv = 0.0f;
    for (int c = 0; c < 32; ++c) { const float d = kx[j * 32 + c] - mn; vv += d * d; }
    mrow[w][j] = mn;
    vrow[w][j] = vv * (1.0f / 32.0f);
  }
  __syncthreads();
  {
    const float g1 = ln1g[j], bb = ln1b[j];
    for (int t = 0; t < 16; ++t) {
      const float r = rsqrtf(vrow[w][t] + 1e-5f);
      kx[t * 32 + j] = (kx[t * 32 + j] - mrow[w][t]) * r * g1 + bb;
    }
  }
  __syncthreads();

  // ---- FFN 32 -> 64 (gelu) -> 32, residual, LN2, residual to raw patch ----
#pragma unroll
  for (int half = 0; half < 2; ++half) {
    const int jc = j + half * 32;
    float wc[32];
    for (int c = 0; c < 32; ++c) wc[c] = w1[c * 64 + jc];
    const float bh1 = b1[jc];
    for (int t = 0; t < 16; ++t) {
      float s = bh1;
#pragma unroll
      for (int c = 0; c < 32; ++c) s += kx[t * 32 + c] * wc[c];
      hb[t * 64 + jc] = gelu_exact(s);
    }
  }
  __syncthreads();
  {
    float wc[64];
    for (int c = 0; c < 64; ++c) wc[c] = w2[c * 32 + j];
    const float bb2 = b2[j];
    for (int t = 0; t < 16; ++t) {
      float s = bb2;
#pragma unroll
      for (int c = 0; c < 64; ++c) s += hb[t * 64 + c] * wc[c];
      vx[t * 32 + j] = kx[t * 32 + j] + s;  // xo pre-LN2
    }
  }
  __syncthreads();
  if (j < 16) {
    float s = 0.0f;
    for (int c = 0; c < 32; ++c) s += vx[j * 32 + c];
    const float mn = s * (1.0f / 32.0f);
    float vv = 0.0f;
    for (int c = 0; c < 32; ++c) { const float d = vx[j * 32 + c] - mn; vv += d * d; }
    mrow[w][j] = mn;
    vrow[w][j] = vv * (1.0f / 32.0f);
  }
  __syncthreads();
  {
    const float g2 = ln2g[j], bb = ln2b[j];
    for (int t = 0; t < 16; ++t) {
      const float r = rsqrtf(vrow[w][t] + 1e-5f);
      const float xo = (vx[t * 32 + j] - mrow[w][t]) * r * g2 + bb;
      const float e = xp[t * 32 + j] + xo;
      z[p * 512 + t * 32 + j] = e;
      zb[p * 512 + t * 32 + j] = (__bf16)e;
    }
  }
}

// ---------------------------------------------------------------------------
// VQ argmin: argmin_k (||c_k||^2 - 2 z.c_k) via WMMA, index-tie-broken.
// One block = 16 rows; wave w covers codes [w*128, w*128+128).
// ---------------------------------------------------------------------------
__global__ __launch_bounds__(256) void vq_argmin_kernel(
    const __bf16* __restrict__ zb, const __bf16* __restrict__ cbb,
    const float* __restrict__ cc, int* __restrict__ idxi,
    float* __restrict__ idxf) {
  __shared__ float bval[8][16];
  __shared__ int bidx[8][16];
  const int l = threadIdx.x & 31;
  const int w = threadIdx.x >> 5;
  const int m = l & 15;
  const int kh = l >> 4;
  const int rowbase = blockIdx.x * 16;
  const __bf16* ap = zb + (size_t)(rowbase + m) * DDIM + kh * 16;

  v8f acc[8] = {};
  for (int kc = 0; kc < DDIM; kc += 32) {
    const v16bf a = *(const v16bf*)(ap + kc);
#pragma unroll
    for (int nt = 0; nt < 8; ++nt) {
      const __bf16* bp = cbb + (size_t)(w * 128 + nt * 16 + m) * DDIM + kh * 16 + kc;
      acc[nt] = wmma_bf16(a, *(const v16bf*)bp, acc[nt]);
    }
  }
  float bv[8];
  int bi[8];
#pragma unroll
  for (int r = 0; r < 8; ++r) { bv[r] = 3.0e38f; bi[r] = 0; }
#pragma unroll
  for (int nt = 0; nt < 8; ++nt) {
    const int n = w * 128 + nt * 16 + m;
    const float ccn = cc[n];
#pragma unroll
    for (int r = 0; r < 8; ++r) {
      const float v = ccn - 2.0f * acc[nt][r];
      if (v < bv[r] || (v == bv[r] && n < bi[r])) { bv[r] = v; bi[r] = n; }
    }
  }
#pragma unroll
  for (int msk = 1; msk <= 8; msk <<= 1) {
#pragma unroll
    for (int r = 0; r < 8; ++r) {
      const float ov = __shfl_xor(bv[r], msk, 32);
      const int oi = __shfl_xor(bi[r], msk, 32);
      if (ov < bv[r] || (ov == bv[r] && oi < bi[r])) { bv[r] = ov; bi[r] = oi; }
    }
  }
  if (m == 0) {
#pragma unroll
    for (int r = 0; r < 8; ++r) {
      bval[w][kh * 8 + r] = bv[r];
      bidx[w][kh * 8 + r] = bi[r];
    }
  }
  __syncthreads();
  if (threadIdx.x < 16) {
    const int t = threadIdx.x;
    float fv = bval[0][t];
    int fi = bidx[0][t];
#pragma unroll
    for (int ww = 1; ww < 8; ++ww) {
      if (bval[ww][t] < fv || (bval[ww][t] == fv && bidx[ww][t] < fi)) {
        fv = bval[ww][t];
        fi = bidx[ww][t];
      }
    }
    idxi[rowbase + t] = fi;
    idxf[rowbase + t] = (float)fi;
  }
}

// ---------------------------------------------------------------------------
// Embed: t = cb[idx] + tf_pos[p], plus per-row (q-z)^2 partial for vq loss.
// ---------------------------------------------------------------------------
__global__ __launch_bounds__(256) void embed_loss_kernel(
    const int* __restrict__ idx, const float* __restrict__ cb,
    const float* __restrict__ z, const float* __restrict__ pos,
    float* __restrict__ t, __bf16* __restrict__ tb, float* __restrict__ lpart) {
  __shared__ float red[256];
  const int row = blockIdx.x;
  const int k = idx[row];
  const int p = row & 255;
  float ls = 0.0f;
  for (int c = threadIdx.x; c < DDIM; c += 256) {
    const float q = cb[(size_t)k * DDIM + c];
    const float d = q - z[(size_t)row * DDIM + c];
    ls += d * d;
    const float tv = q + pos[(size_t)p * DDIM + c];
    t[(size_t)row * DDIM + c] = tv;
    tb[(size_t)row * DDIM + c] = (__bf16)tv;
  }
  red[threadIdx.x] = ls;
  __syncthreads();
  for (int sft = 128; sft > 0; sft >>= 1) {
    if (threadIdx.x < sft) red[threadIdx.x] += red[threadIdx.x + sft];
    __syncthreads();
  }
  if (threadIdx.x == 0) lpart[row] = red[0];
}

__global__ __launch_bounds__(256) void loss_final_kernel(const float* __restrict__ lpart,
                                                         float* __restrict__ out) {
  __shared__ float red[256];
  float s = 0.0f;
  for (int i = threadIdx.x; i < MTOT; i += 256) s += lpart[i];
  red[threadIdx.x] = s;
  __syncthreads();
  for (int sft = 128; sft > 0; sft >>= 1) {
    if (threadIdx.x < sft) red[threadIdx.x] += red[threadIdx.x + sft];
    __syncthreads();
  }
  // vq_loss = q_loss + 0.25*e_loss = 1.25 * mean((q-z)^2)
  if (threadIdx.x == 0) out[0] = red[0] * (1.25f / ((float)MTOT * (float)DDIM));
}

// ---------------------------------------------------------------------------
// Generic bf16 GEMM: C[M][N] = A[M][K] @ Bw[N][K]^T + bias, optional gelu.
// One wave = 16x64 tile; 8 waves per block cover 512 columns.
// ---------------------------------------------------------------------------
__global__ __launch_bounds__(256) void gemm_bf16(
    const __bf16* __restrict__ A, const __bf16* __restrict__ Bw,
    const float* __restrict__ bias, float* __restrict__ Cf,
    __bf16* __restrict__ Cb, int N, int K, int act) {
  const int l = threadIdx.x & 31;
  const int w = threadIdx.x >> 5;
  const int m = l & 15;
  const int kh = l >> 4;
  const int rowbase = blockIdx.x * 16;
  const int colbase = blockIdx.y * 512 + w * 64;
  const __bf16* ap = A + (size_t)(rowbase + m) * K + kh * 16;
  const __bf16* bp0 = Bw + (size_t)(colbase + 0 + m) * K + kh * 16;
  const __bf16* bp1 = Bw + (size_t)(colbase + 16 + m) * K + kh * 16;
  const __bf16* bp2 = Bw + (size_t)(colbase + 32 + m) * K + kh * 16;
  const __bf16* bp3 = Bw + (size_t)(colbase + 48 + m) * K + kh * 16;
  v8f acc0 = {}, acc1 = {}, acc2 = {}, acc3 = {};
  for (int kc = 0; kc < K; kc += 32) {
    const v16bf a = *(const v16bf*)(ap + kc);
    acc0 = wmma_bf16(a, *(const v16bf*)(bp0 + kc), acc0);
    acc1 = wmma_bf16(a, *(const v16bf*)(bp1 + kc), acc1);
    acc2 = wmma_bf16(a, *(const v16bf*)(bp2 + kc), acc2);
    acc3 = wmma_bf16(a, *(const v16bf*)(bp3 + kc), acc3);
  }
  const int row0 = rowbase + kh * 8;
  v8f accs[4] = {acc0, acc1, acc2, acc3};
#pragma unroll
  for (int t = 0; t < 4; ++t) {
    const int col = colbase + t * 16 + m;
    const float bs = bias ? bias[col] : 0.0f;
#pragma unroll
    for (int r = 0; r < 8; ++r) {
      float val = accs[t][r] + bs;
      if (act == 1) val = gelu_exact(val);
      const size_t o = (size_t)(row0 + r) * N + col;
      if (Cf) Cf[o] = val;
      if (Cb) Cb[o] = (__bf16)val;
    }
  }
}

// ---------------------------------------------------------------------------
// V transpose: qkv[..][1024 + h*64 + d] -> vT[b][h][d][p]
// ---------------------------------------------------------------------------
__global__ __launch_bounds__(256) void vtrans_kernel(const __bf16* __restrict__ qkv,
                                                     __bf16* __restrict__ vT) {
  const size_t i = (size_t)blockIdx.x * 256 + threadIdx.x;
  const int p = (int)(i & 255);
  const size_t r = i >> 8;
  const int d = (int)(r & 63);
  const size_t bh = r >> 6;
  const int h = (int)(bh & 7);
  const size_t b = bh >> 3;
  vT[i] = qkv[(size_t)(b * 256 + p) * NQKV + 1024 + h * 64 + d];
}

// ---------------------------------------------------------------------------
// Causal attention: one wave per (batch, head, 16-query block).
// Pass 1: exact row-max via WMMA score tiles. Pass 2: probs -> LDS transpose
// -> WMMA P@V against pre-transposed V.
// ---------------------------------------------------------------------------
__global__ __launch_bounds__(256) void attn_kernel(const __bf16* __restrict__ qkv,
                                                   const __bf16* __restrict__ vT,
                                                   __bf16* __restrict__ outb) {
  __shared__ __align__(32) __bf16 ptile[8][16][32];
  const int l = threadIdx.x & 31;
  const int w = threadIdx.x >> 5;
  const int m = l & 15;
  const int kh = l >> 4;
  const int gid = blockIdx.x * 8 + w;
  const int b = gid >> 7;
  const int h = (gid >> 4) & 7;
  const int qb = gid & 15;
  const int qrow = b * 256 + qb * 16;

  const __bf16* qp = qkv + (size_t)(qrow + m) * NQKV + h * 64 + kh * 16;
  const v16bf qa0 = *(const v16bf*)(qp);
  const v16bf qa1 = *(const v16bf*)(qp + 32);

  float rmax[8];
#pragma unroll
  for (int r = 0; r < 8; ++r) rmax[r] = -3.0e38f;

  for (int jt = 0; jt <= qb; ++jt) {
    const __bf16* kp = qkv + (size_t)(b * 256 + jt * 16 + m) * NQKV + 512 + h * 64 + kh * 16;
    v8f acc = {};
    acc = wmma_bf16(qa0, *(const v16bf*)(kp), acc);
    acc = wmma_bf16(qa1, *(const v16bf*)(kp + 32), acc);
    const bool diag = (jt == qb);
#pragma unroll
    for (int r = 0; r < 8; ++r) {
      float v = acc[r] * 0.125f;  // 1/sqrt(64)
      if (diag && (m > kh * 8 + r)) v = -3.0e38f;
      rmax[r] = fmaxf(rmax[r], v);
    }
  }
#pragma unroll
  for (int msk = 1; msk <= 8; msk <<= 1)
#pragma unroll
    for (int r = 0; r < 8; ++r)
      rmax[r] = fmaxf(rmax[r], __shfl_xor(rmax[r], msk, 32));

  float psum[8] = {0, 0, 0, 0, 0, 0, 0, 0};
  v8f oacc[4] = {};
  const int nch = (qb + 2) >> 1;  // 32-key chunks covering (qb+1)*16 keys
  for (int kc = 0; kc < nch; ++kc) {
#pragma unroll
    for (int sub = 0; sub < 2; ++sub) {
      const int kb0 = kc * 32 + sub * 16;
      if (kb0 <= qb * 16) {
        const __bf16* kp = qkv + (size_t)(b * 256 + kb0 + m) * NQKV + 512 + h * 64 + kh * 16;
        v8f acc = {};
        acc = wmma_bf16(qa0, *(const v16bf*)(kp), acc);
        acc = wmma_bf16(qa1, *(const v16bf*)(kp + 32), acc);
        const bool diag = (kb0 == qb * 16);
#pragma unroll
        for (int r = 0; r < 8; ++r) {
          const float v = acc[r] * 0.125f;
          const float pv = (diag && (m > kh * 8 + r)) ? 0.0f : __expf(v - rmax[r]);
          psum[r] += pv;
          ptile[w][kh * 8 + r][sub * 16 + m] = (__bf16)pv;
        }
      } else {
#pragma unroll
        for (int r = 0; r < 8; ++r) ptile[w][kh * 8 + r][sub * 16 + m] = (__bf16)0.0f;
      }
    }
    __builtin_amdgcn_wave_barrier();  // LDS C-layout -> A-layout handoff
    const v16bf pa = *(const v16bf*)(&ptile[w][m][kh * 16]);
#pragma unroll
    for (int nt = 0; nt < 4; ++nt) {
      const __bf16* vp = vT + ((size_t)(b * 8 + h) * 64 + nt * 16 + m) * 256 + kc * 32 + kh * 16;
      oacc[nt] = wmma_bf16(pa, *(const v16bf*)(vp), oacc[nt]);
    }
    __builtin_amdgcn_wave_barrier();
  }
#pragma unroll
  for (int msk = 1; msk <= 8; msk <<= 1)
#pragma unroll
    for (int r = 0; r < 8; ++r) psum[r] += __shfl_xor(psum[r], msk, 32);

#pragma unroll
  for (int nt = 0; nt < 4; ++nt)
#pragma unroll
    for (int r = 0; r < 8; ++r) {
      const float val = oacc[nt][r] / psum[r];
      outb[(size_t)(qrow + kh * 8 + r) * DDIM + h * 64 + nt * 16 + m] = (__bf16)val;
    }
}

// ---------------------------------------------------------------------------
// Residual-add + LayerNorm (one wave per 512-wide row). add==nullptr -> plain LN.
// ---------------------------------------------------------------------------
__global__ __launch_bounds__(256) void ln_res_kernel(
    const float* __restrict__ xin, const float* __restrict__ add,
    const float* __restrict__ g, const float* __restrict__ bta,
    float* __restrict__ of, __bf16* __restrict__ ob) {
  const int l = threadIdx.x & 31;
  const int w = threadIdx.x >> 5;
  const size_t row = (size_t)blockIdx.x * 8 + w;
  const float* xr = xin + row * DDIM;
  float v[16];
  float s = 0.0f;
#pragma unroll
  for (int i = 0; i < 16; ++i) {
    float t = xr[l + 32 * i];
    if (add) t += add[row * DDIM + l + 32 * i];
    v[i] = t;
    s += t;
  }
#pragma unroll
  for (int msk = 1; msk <= 16; msk <<= 1) s += __shfl_xor(s, msk, 32);
  const float mean = s * (1.0f / 512.0f);
  float sq = 0.0f;
#pragma unroll
  for (int i = 0; i < 16; ++i) { const float d = v[i] - mean; sq += d * d; }
#pragma unroll
  for (int msk = 1; msk <= 16; msk <<= 1) sq += __shfl_xor(sq, msk, 32);
  const float rstd = rsqrtf(sq * (1.0f / 512.0f) + 1e-5f);
#pragma unroll
  for (int i = 0; i < 16; ++i) {
    const int c = l + 32 * i;
    const float o = (v[i] - mean) * rstd * g[c] + bta[c];
    of[row * DDIM + c] = o;
    if (ob) ob[row * DDIM + c] = (__bf16)o;
  }
}

// ---------------------------------------------------------------------------
extern "C" void kernel_launch(void* const* d_in, const int* in_sizes, int n_in,
                              void* d_out, int out_size, void* d_ws, size_t ws_size,
                              hipStream_t stream) {
  (void)in_sizes; (void)n_in; (void)out_size; (void)ws_size;
  const float* x       = (const float*)d_in[0];
  const float* pa_lq   = (const float*)d_in[1];
  const float* pa_pos  = (const float*)d_in[2];
  const float* pa_win  = (const float*)d_in[3];
  const float* pa_bin  = (const float*)d_in[4];
  const float* pa_wout = (const float*)d_in[5];
  const float* pa_bout = (const float*)d_in[6];
  const float* pa_ln1g = (const float*)d_in[7];
  const float* pa_ln1b = (const float*)d_in[8];
  const float* pa_ln2g = (const float*)d_in[9];
  const float* pa_ln2b = (const float*)d_in[10];
  const float* pa_w1   = (const float*)d_in[11];
  const float* pa_b1   = (const float*)d_in[12];
  const float* pa_w2   = (const float*)d_in[13];
  const float* pa_b2   = (const float*)d_in[14];
  const float* cb      = (const float*)d_in[15];
  const float* tf_pos  = (const float*)d_in[16];
  const float* tf_win  = (const float*)d_in[17];
  const float* tf_bin  = (const float*)d_in[18];
  const float* tf_wout = (const float*)d_in[19];
  const float* tf_bout = (const float*)d_in[20];
  const float* tf_ln1g = (const float*)d_in[21];
  const float* tf_ln1b = (const float*)d_in[22];
  const float* tf_ln2g = (const float*)d_in[23];
  const float* tf_ln2b = (const float*)d_in[24];
  const float* tf_w1   = (const float*)d_in[25];
  const float* tf_b1   = (const float*)d_in[26];
  const float* tf_w2   = (const float*)d_in[27];
  const float* tf_b2   = (const float*)d_in[28];
  const float* fin_g   = (const float*)d_in[29];
  const float* fin_b   = (const float*)d_in[30];

  char* ws = (char*)d_ws;
  size_t off = 0;
  auto alloc = [&](size_t bytes) -> void* {
    void* pp = ws + off;
    off += (bytes + 255) & ~(size_t)255;
    return pp;
  };
  float*  z_f   = (float*) alloc((size_t)MTOT * DDIM * 4);
  __bf16* z_b   = (__bf16*)alloc((size_t)MTOT * DDIM * 2);
  __bf16* cb_b  = (__bf16*)alloc((size_t)NCODE * DDIM * 2);
  float*  cc_v  = (float*) alloc((size_t)NCODE * 4);
  int*    idx_v = (int*)   alloc((size_t)MTOT * 4);
  float*  lpart = (float*) alloc((size_t)MTOT * 4);
  float*  t_f   = (float*) alloc((size_t)MTOT * DDIM * 4);
  __bf16* t_b   = (__bf16*)alloc((size_t)MTOT * DDIM * 2);
  float*  abuf  = (float*) alloc((size_t)MTOT * DDIM * 4);
  __bf16* qkvb  = (__bf16*)alloc((size_t)MTOT * NQKV * 2);
  __bf16* vTb   = (__bf16*)alloc((size_t)MTOT * DDIM * 2);
  __bf16* attb  = (__bf16*)alloc((size_t)MTOT * DDIM * 2);
  __bf16* f1b   = (__bf16*)alloc((size_t)MTOT * DFFN * 2);
  __bf16* winb  = (__bf16*)alloc((size_t)4 * NQKV * DDIM * 2);
  __bf16* woutb = (__bf16*)alloc((size_t)4 * DDIM * DDIM * 2);
  __bf16* w1tb  = (__bf16*)alloc((size_t)4 * DFFN * DDIM * 2);
  __bf16* w2tb  = (__bf16*)alloc((size_t)4 * DDIM * DFFN * 2);

  float* y_out = (float*)d_out;
  float* loss_out = y_out + (size_t)MTOT * DDIM;
  float* idx_out = loss_out + 1;

  // one-time conversions (per launch; deterministic, no cached state)
  cvt_kernel<<<(NCODE * DDIM + 255) / 256, 256, 0, stream>>>(cb, cb_b, NCODE * DDIM);
  cc_kernel<<<NCODE / 8, 256, 0, stream>>>(cb, cc_v);
  cvt_kernel<<<(4 * NQKV * DDIM + 255) / 256, 256, 0, stream>>>(tf_win, winb, 4 * NQKV * DDIM);
  cvt_kernel<<<(4 * DDIM * DDIM + 255) / 256, 256, 0, stream>>>(tf_wout, woutb, 4 * DDIM * DDIM);
  cvt_tr_kernel<<<(4 * DDIM * DFFN + 255) / 256, 256, 0, stream>>>(tf_w1, w1tb, DDIM, DFFN, 4 * DDIM * DFFN);
  cvt_tr_kernel<<<(4 * DFFN * DDIM + 255) / 256, 256, 0, stream>>>(tf_w2, w2tb, DFFN, DDIM, 4 * DFFN * DDIM);

  patch_encode_kernel<<<MTOT / PWAVES, PWAVES * 32, 0, stream>>>(
      x, pa_lq, pa_pos, pa_win, pa_bin, pa_wout, pa_bout,
      pa_ln1g, pa_ln1b, pa_ln2g, pa_ln2b, pa_w1, pa_b1, pa_w2, pa_b2, z_f, z_b);

  vq_argmin_kernel<<<MTOT / 16, 256, 0, stream>>>(z_b, cb_b, cc_v, idx_v, idx_out);
  embed_loss_kernel<<<MTOT, 256, 0, stream>>>(idx_v, cb, z_f, tf_pos, t_f, t_b, lpart);
  loss_final_kernel<<<1, 256, 0, stream>>>(lpart, loss_out);

  for (int i = 0; i < 4; ++i) {
    gemm_bf16<<<dim3(MTOT / 16, NQKV / 512), 256, 0, stream>>>(
        t_b, winb + (size_t)i * NQKV * DDIM, tf_bin + i * NQKV, nullptr, qkvb, NQKV, DDIM, 0);
    vtrans_kernel<<<(MTOT * DDIM) / 256, 256, 0, stream>>>(qkvb, vTb);
    attn_kernel<<<(64 * 8 * 16) / 8, 256, 0, stream>>>(qkvb, vTb, attb);
    gemm_bf16<<<dim3(MTOT / 16, 1), 256, 0, stream>>>(
        attb, woutb + (size_t)i * DDIM * DDIM, tf_bout + i * DDIM, abuf, nullptr, DDIM, DDIM, 0);
    ln_res_kernel<<<MTOT / 8, 256, 0, stream>>>(t_f, abuf, tf_ln1g + i * DDIM, tf_ln1b + i * DDIM, t_f, t_b);
    gemm_bf16<<<dim3(MTOT / 16, DFFN / 512), 256, 0, stream>>>(
        t_b, w1tb + (size_t)i * DFFN * DDIM, tf_b1 + i * DFFN, nullptr, f1b, DFFN, DDIM, 1);
    gemm_bf16<<<dim3(MTOT / 16, 1), 256, 0, stream>>>(
        f1b, w2tb + (size_t)i * DDIM * DFFN, tf_b2 + i * DDIM, abuf, nullptr, DDIM, DFFN, 0);
    ln_res_kernel<<<MTOT / 8, 256, 0, stream>>>(t_f, abuf, tf_ln2g + i * DDIM, tf_ln2b + i * DDIM, t_f, t_b);
  }
  ln_res_kernel<<<MTOT / 8, 256, 0, stream>>>(t_f, nullptr, fin_g, fin_b, y_out, nullptr);
}
